// BiLSTM_58351425683848
// MI455X (gfx1250) — matrix operand order repaired and checked by reading, not compiled, not run on hardware
//
#include <hip/hip_runtime.h>

#define S_LEN 256
#define HID   1024
#define EPSB  1e-5f

typedef __attribute__((ext_vector_type(16))) __bf16 v16bf;
typedef __attribute__((ext_vector_type(8)))  float  v8f;

struct __attribute__((aligned(16))) U4 { unsigned int x, y, z, w; };

union Frag16 {
    U4           q[2];
    unsigned int u[8];
    v16bf        v;
};

__device__ __forceinline__ unsigned short f2bf_rne(float f) {
    unsigned int u = __float_as_uint(f);
    u += 0x7FFFu + ((u >> 16) & 1u);      // round-to-nearest-even
    return (unsigned short)(u >> 16);
}

__device__ __forceinline__ float sigmoid_f(float x) {
    return 1.0f / (1.0f + __expf(-x));
}

// ---------------------------------------------------------------------------
// C[m,n] = sigmoid( sum_k (A1[m,k]+A2[m,k]) * W[n,k] + bias[n] )
// A1,A2: [256,1024] f32.  W: [1024,1024] bf16 row-major (N rows, K contiguous).
// Block = 256 threads (8 waves). Block tile: 16 (M) x 256 (N); wave tile 16x32.
// grid.x = 16 (M tiles), grid.y = 4 (N groups of 256).
// ---------------------------------------------------------------------------
__device__ __forceinline__ void gemm_sig_body(
    const float* __restrict__ A1, const float* __restrict__ A2,
    const unsigned short* __restrict__ Wb, const float* __restrict__ bias,
    float* __restrict__ Out, int mblk, int nblk)
{
    constexpr int RS = HID + 4;                 // padded LDS row stride (bf16)
    __shared__ unsigned short As[16 * RS];

    const int tid = threadIdx.x;
    const int m0  = mblk * 16;

    // Stage A = A1 + A2 (rows m0..m0+15) into LDS as bf16.
    for (int i = tid; i < 16 * HID; i += 256) {
        int r = i >> 10;
        int c = i & (HID - 1);
        float v = A1[(size_t)(m0 + r) * HID + c] + A2[(size_t)(m0 + r) * HID + c];
        As[r * RS + c] = f2bf_rne(v);
    }
    __syncthreads();

    const int w     = tid >> 5;
    const int lane  = tid & 31;
    const int mrow  = lane & 15;
    const int half  = lane >> 4;
    const int nbase = nblk * 256 + w * 32;
    const int ncol  = nbase + (lane & 15);

    v8f acc0 = {0.f,0.f,0.f,0.f,0.f,0.f,0.f,0.f};
    v8f acc1 = {0.f,0.f,0.f,0.f,0.f,0.f,0.f,0.f};

    for (int k0 = 0; k0 < HID; k0 += 32) {
        // A fragment: lane holds row m, K = 16*half + 2j+p (contiguous 16 bf16)
        Frag16 a;
#pragma unroll
        for (int j = 0; j < 8; ++j) {
            int k = k0 + (half << 4) + (j << 1);
            a.u[j] = *(const unsigned int*)&As[mrow * RS + k];
        }
        // B fragments: lane holds column n, K = 16*half + 2j+p (contiguous 16 bf16)
        const size_t koff = (size_t)k0 + (size_t)(half << 4);
        const U4* wp0 = (const U4*)(Wb + (size_t)(ncol     ) * HID + koff);
        const U4* wp1 = (const U4*)(Wb + (size_t)(ncol + 16) * HID + koff);
        Frag16 b0, b1;
        b0.q[0] = wp0[0]; b0.q[1] = wp0[1];
        b1.q[0] = wp1[0]; b1.q[1] = wp1[1];

        acc0 = __builtin_amdgcn_wmma_f32_16x16x32_bf16(
                   false, a.v, false, b0.v, (short)0, acc0, false, false);
        acc1 = __builtin_amdgcn_wmma_f32_16x16x32_bf16(
                   false, a.v, false, b1.v, (short)0, acc1, false, false);
    }

    // Epilogue: C layout m = m0 + 8*half + j, n = nbase + 16*i + (lane&15)
    v8f accs[2] = {acc0, acc1};
#pragma unroll
    for (int i = 0; i < 2; ++i) {
        int n = nbase + i * 16 + (lane & 15);
        float bv = bias[n];
#pragma unroll
        for (int j = 0; j < 8; ++j) {
            int m = m0 + (half << 3) + j;
            Out[(size_t)m * HID + n] = sigmoid_f(accs[i][j] + bv);
        }
    }
}

__global__ void k_gemm1(const float* __restrict__ inp, const float* __restrict__ hf,
                        const unsigned short* __restrict__ Wb1,
                        const float* __restrict__ b1, float* __restrict__ x1)
{
    gemm_sig_body(inp, hf, Wb1, b1, x1, blockIdx.x, blockIdx.y);
}

__global__ void k_gemm23(const float* __restrict__ x1, const float* __restrict__ hf,
                         const float* __restrict__ hb,
                         const unsigned short* __restrict__ Wb2,
                         const float* __restrict__ b2, float* __restrict__ hb2,
                         const unsigned short* __restrict__ Wb3,
                         const float* __restrict__ b3, float* __restrict__ hf2)
{
    if (blockIdx.z == 0)
        gemm_sig_body(hb, x1, Wb2, b2, hb2, blockIdx.x, blockIdx.y);  // hb2
    else
        gemm_sig_body(x1, hf, Wb3, b3, hf2, blockIdx.x, blockIdx.y);  // hf2
}

// Training-mode BatchNorm1d over the 256-row axis, both streams in one launch.
// grid.x = 8 (blocks 0..3 -> forward stream, 4..7 -> backward), block = 256.
__global__ void k_bn(const float* __restrict__ hf2, const float* __restrict__ gf,
                     const float* __restrict__ bf, float* __restrict__ hf,
                     const float* __restrict__ hb2, const float* __restrict__ gb,
                     const float* __restrict__ bb, float* __restrict__ hb)
{
    const int col = (blockIdx.x & 3) * 256 + threadIdx.x;
    const float* src; const float* g; const float* be; float* dst;
    if (blockIdx.x < 4) { src = hf2; g = gf; be = bf; dst = hf; }
    else                { src = hb2; g = gb; be = bb; dst = hb; }

    float s = 0.f, s2 = 0.f;
    for (int r = 0; r < S_LEN; ++r) {
        float v = src[(size_t)r * HID + col];
        s += v; s2 += v * v;
    }
    const float mean = s  * (1.0f / S_LEN);
    const float var  = s2 * (1.0f / S_LEN) - mean * mean;   // biased, matches ref
    const float rstd = rsqrtf(var + EPSB);
    const float gg = g[col], bb_ = be[col];
    for (int r = 0; r < S_LEN; ++r) {
        float v = src[(size_t)r * HID + col];
        dst[(size_t)r * HID + col] = (v - mean) * rstd * gg + bb_;
    }
}

__global__ void k_cvt(const float* __restrict__ W1, const float* __restrict__ W2,
                      const float* __restrict__ W3,
                      unsigned short* __restrict__ o1, unsigned short* __restrict__ o2,
                      unsigned short* __restrict__ o3)
{
    const int z = blockIdx.y;
    const float* src = (z == 0) ? W1 : (z == 1) ? W2 : W3;
    unsigned short* dst = (z == 0) ? o1 : (z == 1) ? o2 : o3;
    const int i = blockIdx.x * 256 + threadIdx.x;
    dst[i] = f2bf_rne(src[i]);
}

__global__ void k_zero(float* __restrict__ hf, float* __restrict__ hb)
{
    const int i = blockIdx.x * 256 + threadIdx.x;
    hf[i] = 0.f;
    hb[i] = 0.f;
}

__global__ void k_out(const float* __restrict__ hf2, const float* __restrict__ hb2,
                      float* __restrict__ out)
{
    const int i = blockIdx.x * 256 + threadIdx.x;
    out[i] = 0.5f * (hf2[i] + hb2[i]);
}

extern "C" void kernel_launch(void* const* d_in, const int* in_sizes, int n_in,
                              void* d_out, int out_size, void* d_ws, size_t ws_size,
                              hipStream_t stream)
{
    (void)in_sizes; (void)n_in; (void)out_size; (void)ws_size;

    const float* inp = (const float*)d_in[0];
    const float* W1  = (const float*)d_in[1];
    const float* b1  = (const float*)d_in[2];
    const float* W2  = (const float*)d_in[3];
    const float* b2  = (const float*)d_in[4];
    const float* W3  = (const float*)d_in[5];
    const float* b3  = (const float*)d_in[6];
    // d_in[7..16]: W4/b4 + LSTM weights — dead code w.r.t. the returned output.
    const float* gf  = (const float*)d_in[17];
    const float* bf  = (const float*)d_in[18];
    const float* gb  = (const float*)d_in[19];
    const float* bb  = (const float*)d_in[20];

    char* ws = (char*)d_ws;
    size_t off = 0;
    auto carve = [&](size_t bytes) -> void* {
        void* p = ws + off;
        off += (bytes + 255) & ~(size_t)255;
        return p;
    };
    unsigned short* Wb1 = (unsigned short*)carve((size_t)HID * HID * 2);
    unsigned short* Wb2 = (unsigned short*)carve((size_t)HID * HID * 2);
    unsigned short* Wb3 = (unsigned short*)carve((size_t)HID * HID * 2);
    float* hf  = (float*)carve((size_t)S_LEN * HID * 4);
    float* hb  = (float*)carve((size_t)S_LEN * HID * 4);
    float* x1  = (float*)carve((size_t)S_LEN * HID * 4);
    float* hf2 = (float*)carve((size_t)S_LEN * HID * 4);
    float* hb2 = (float*)carve((size_t)S_LEN * HID * 4);

    // One-time prep: bf16 weights (live in L2 for the whole run) + zero state.
    k_cvt<<<dim3(HID * HID / 256, 3), 256, 0, stream>>>(W1, W2, W3, Wb1, Wb2, Wb3);
    k_zero<<<dim3(S_LEN * HID / 256), 256, 0, stream>>>(hf, hb);

    const dim3 ggrid(S_LEN / 16, HID / 256);       // 16 x 4
    const dim3 ggrid2(S_LEN / 16, HID / 256, 2);   // fused hb2/hf2

    for (int t = 0; t < S_LEN; ++t) {
        k_gemm1 <<<ggrid,  256, 0, stream>>>(inp, hf, Wb1, b1, x1);
        k_gemm23<<<ggrid2, 256, 0, stream>>>(x1, hf, hb, Wb2, b2, hb2, Wb3, b3, hf2);
        if (t < S_LEN - 1)   // last step's BN output is never consumed
            k_bn<<<dim3(8), 256, 0, stream>>>(hf2, gf, bf, hf, hb2, gb, bb, hb);
    }
    k_out<<<dim3(S_LEN * HID / 256), 256, 0, stream>>>(hf2, hb2, (float*)d_out);
}
